// ProjectedAdaptiveLogSoftmax_76270029243071
// MI455X (gfx1250) — compile-verified, matrix operand order checked by reading, not run on hardware
//
#include <hip/hip_runtime.h>
#include <cmath>

typedef __bf16 v16bf __attribute__((ext_vector_type(16)));
typedef float  v8f   __attribute__((ext_vector_type(8)));
typedef unsigned int   u32;
typedef unsigned short u16;

#define D_EMBED   1024
#define M_ROWS    64            // rows per block
#define RG        4             // 16-row subtiles per wave (k-outer, shared B)
#define WAVES     8
#define TILE_COLS 128           // 8 waves * 16 cols
#define CHUNKS    8             // vocab split for occupancy
#define A_STRIDE  (D_EMBED + 8) // bf16 elems per LDS row (pad -> conflict-floor b128)
#define NEG_INF   (-3.0e38f)
#define NVOCAB_EXT 50002        // 50000 vocab rows + 2 cluster rows
#define N_TOK     4096

// -------- workspace layout (bytes); total ~196.2 MiB --------
#define SZ_PLANE ((size_t)NVOCAB_EXT * D_EMBED * 2)
#define OFF_WHI  ((size_t)0)
#define OFF_WLO  (OFF_WHI + SZ_PLANE)
#define SZ_PART  ((size_t)3 * CHUNKS * N_TOK * 4)
#define OFF_PM   (OFF_WLO + SZ_PLANE)
#define OFF_PS   (OFF_PM + SZ_PART)
#define OFF_TGT  (OFF_PS + SZ_PART)
#define OFF_CLUS (OFF_TGT + (size_t)3 * N_TOK * 4)

__device__ __forceinline__ u16 f32_to_bf16_rne(float f) {
    u32 u = __float_as_uint(f);
    u32 r = u + 0x7fffu + ((u >> 16) & 1u);
    return (u16)(r >> 16);
}

// Split f32 weight (+appended cluster rows) into bf16 hi/lo planes.
// hi = truncate-to-top16 (so f - hi is exact in f32), lo = RNE(residual).
__global__ void convert_weight_split(const float* __restrict__ weight,
                                     const float* __restrict__ cw,
                                     u16* __restrict__ whi, u16* __restrict__ wlo) {
    const size_t npairs = (size_t)NVOCAB_EXT * (D_EMBED / 2);
    for (size_t p = (size_t)blockIdx.x * blockDim.x + threadIdx.x; p < npairs;
         p += (size_t)gridDim.x * blockDim.x) {
        size_t i   = p * 2;
        size_t row = i >> 10;
        int    k   = (int)(i & (D_EMBED - 1));
        const float* src = (row < 50000) ? (weight + i)
                                         : (cw + (row - 50000) * D_EMBED + k);
        float f0 = src[0], f1 = src[1];
        u32 u0 = __float_as_uint(f0), u1 = __float_as_uint(f1);
        float lo0 = f0 - __uint_as_float(u0 & 0xffff0000u);
        float lo1 = f1 - __uint_as_float(u1 & 0xffff0000u);
        *(u32*)(whi + i) = (u0 >> 16) | (u1 & 0xffff0000u);
        *(u32*)(wlo + i) = (u32)f32_to_bf16_rne(lo0) | ((u32)f32_to_bf16_rne(lo1) << 16);
    }
}

__global__ void init_gather(float* __restrict__ gtgt, float* __restrict__ gclus) {
    int i = blockIdx.x * blockDim.x + threadIdx.x;
    if (i < 3 * N_TOK) gtgt[i] = NEG_INF;
    if (i < 2 * N_TOK) gclus[i] = 0.f;
}

union Frag { uint4 q[2]; v16bf v; };

__global__ __launch_bounds__(256)
void adaptive_lsm_gemm(const float* __restrict__ hidden,
                       const int*   __restrict__ target,
                       const float* __restrict__ bias,
                       const float* __restrict__ cluster_bias,
                       const u16* __restrict__ whi, const u16* __restrict__ wlo,
                       float* __restrict__ pm, float* __restrict__ ps,
                       float* __restrict__ gtgt, float* __restrict__ gclus)
{
    extern __shared__ char smemraw[];
    u16*   a_hi  = (u16*)smemraw;                    // [M_ROWS][A_STRIDE]
    u16*   a_lo  = a_hi + M_ROWS * A_STRIDE;         // [M_ROWS][A_STRIDE]
    int*   s_tid = (int*)(a_lo + M_ROWS * A_STRIDE); // [M_ROWS]
    float* s_m   = (float*)(s_tid + M_ROWS);         // [WAVES][M_ROWS]
    float* s_s   = s_m + WAVES * M_ROWS;             // [WAVES][M_ROWS]

    const int tid  = threadIdx.x;
    const int wave = tid >> 5, lane = tid & 31, half = lane >> 4, lh = lane & 15;
    const int rowBase = blockIdx.x * M_ROWS;
    const int chunk   = blockIdx.y;

    // Stage hidden tile, splitting into bf16 hi/lo planes in LDS.
    for (int p = tid; p < M_ROWS * (D_EMBED / 2); p += 256) {
        int m = p >> 9;
        int k = (p & 511) * 2;
        const float* src = hidden + (size_t)(rowBase + m) * D_EMBED + k;
        float f0 = src[0], f1 = src[1];
        u32 u0 = __float_as_uint(f0), u1 = __float_as_uint(f1);
        float lo0 = f0 - __uint_as_float(u0 & 0xffff0000u);
        float lo1 = f1 - __uint_as_float(u1 & 0xffff0000u);
        *(u32*)(a_hi + m * A_STRIDE + k) = (u0 >> 16) | (u1 & 0xffff0000u);
        *(u32*)(a_lo + m * A_STRIDE + k) =
            (u32)f32_to_bf16_rne(lo0) | ((u32)f32_to_bf16_rne(lo1) << 16);
    }
    if (tid < M_ROWS) s_tid[tid] = target[rowBase + tid];
    __syncthreads();

    const v8f vzero = {0.f, 0.f, 0.f, 0.f, 0.f, 0.f, 0.f, 0.f};

    for (int seg = 0; seg < 3; ++seg) {
        const int ncols  = (seg == 0) ? 20002 : ((seg == 1) ? 20000 : 10000);
        const int rowoff = (seg == 0) ? 0     : ((seg == 1) ? 20000 : 40000);
        const int ntiles = (ncols + TILE_COLS - 1) / TILE_COLS;

        float run_m[RG * 8], run_s[RG * 8];
        #pragma unroll
        for (int i = 0; i < RG * 8; ++i) { run_m[i] = NEG_INF; run_s[i] = 0.f; }

        for (int t = chunk; t < ntiles; t += CHUNKS) {
            const int  col   = t * TILE_COLS + wave * 16 + lh;  // this lane's N
            const bool valid = (col < ncols);
            int wIdx; float bval;
            if (seg == 0) {
                if (col < 20000) { wIdx = col; bval = bias[col]; }
                else { int c2 = valid ? (col - 20000) : 0;
                       wIdx = 50000 + c2; bval = cluster_bias[c2]; }
            } else {
                int c = valid ? col : 0;
                wIdx = rowoff + c; bval = bias[wIdx];
            }
            // B frag (bf16 32x16): lane(half,lh): N=lh, chunks K=[kb+8h,+8) and K=[kb+16+8h,+8)
            const u16* bh = whi + (size_t)wIdx * D_EMBED + half * 8;
            const u16* bl = wlo + (size_t)wIdx * D_EMBED + half * 8;

            v8f acc[RG];
            #pragma unroll
            for (int g = 0; g < RG; ++g) acc[g] = vzero;

            // double-buffered B: next k-step's global loads issue before this
            // step's WMMA chain so waits can be partial, not loadcnt==0.
            Frag bhf, blf, bhn, bln;
            bhf.q[0] = *(const uint4*)(bh);      bhf.q[1] = *(const uint4*)(bh + 16);
            blf.q[0] = *(const uint4*)(bl);      blf.q[1] = *(const uint4*)(bl + 16);

            for (int kb = 0; kb < D_EMBED; kb += 32) {
                const int kn = (kb + 32 < D_EMBED) ? (kb + 32) : 0;
                bhn.q[0] = *(const uint4*)(bh + kn);
                bhn.q[1] = *(const uint4*)(bh + kn + 16);
                bln.q[0] = *(const uint4*)(bl + kn);
                bln.q[1] = *(const uint4*)(bl + kn + 16);

                #pragma unroll
                for (int g = 0; g < RG; ++g) {
                    const u16* ap  = a_hi + (g * 16 + lh) * A_STRIDE + kb + half * 8;
                    const u16* alp = a_lo + (g * 16 + lh) * A_STRIDE + kb + half * 8;
                    Frag ahf, alf;
                    ahf.q[0] = *(const uint4*)(ap);       ahf.q[1] = *(const uint4*)(ap + 16);
                    alf.q[0] = *(const uint4*)(alp);      alf.q[1] = *(const uint4*)(alp + 16);
                    // bf16x3 split product: hi*hi + hi*lo + lo*hi (f32 accumulate)
                    acc[g] = __builtin_amdgcn_wmma_f32_16x16x32_bf16(
                        false, ahf.v, false, bhf.v, (short)0, acc[g], false, false);
                    acc[g] = __builtin_amdgcn_wmma_f32_16x16x32_bf16(
                        false, ahf.v, false, blf.v, (short)0, acc[g], false, false);
                    acc[g] = __builtin_amdgcn_wmma_f32_16x16x32_bf16(
                        false, alf.v, false, bhf.v, (short)0, acc[g], false, false);
                }
                bhf = bhn; blf = bln;
            }

            // bias + mask + gather + online logsumexp update
            #pragma unroll
            for (int g = 0; g < RG; ++g) {
                #pragma unroll
                for (int r = 0; r < 8; ++r) {
                    float v = valid ? (acc[g][r] + bval) : NEG_INF;
                    const int lrow = g * 16 + r + half * 8; // C layout: r->M, half->+8
                    const int grow = rowBase + lrow;
                    const int tt   = s_tid[lrow];
                    if (seg == 0) {
                        if (tt < 20000 && col == tt) gtgt[grow] = v;
                        if (col == 20000) gclus[grow] = v;            // head_lp[:,-2]
                        if (col == 20001) gclus[N_TOK + grow] = v;    // head_lp[:,-1]
                    } else if (seg == 1) {
                        if (tt >= 20000 && tt < 40000 && col == tt - 20000)
                            gtgt[N_TOK + grow] = v;
                    } else {
                        if (tt >= 40000 && col == tt - 40000)
                            gtgt[2 * N_TOK + grow] = v;
                    }
                    const int si = g * 8 + r;
                    float mn = fmaxf(run_m[si], v);
                    run_s[si] = run_s[si] * __expf(run_m[si] - mn) + __expf(v - mn);
                    run_m[si] = mn;
                }
            }
        }

        // reduce across the 16 lanes of each half, then across waves via LDS
        #pragma unroll
        for (int g = 0; g < RG; ++g) {
            #pragma unroll
            for (int r = 0; r < 8; ++r) {
                const int si = g * 8 + r;
                float m = run_m[si], s = run_s[si];
                #pragma unroll
                for (int off = 1; off <= 8; off <<= 1) {
                    float mo = __shfl_xor(m, off, 32);
                    float so = __shfl_xor(s, off, 32);
                    float mn = fmaxf(m, mo);
                    s = s * __expf(m - mn) + so * __expf(mo - mn);
                    m = mn;
                }
                if (lh == 0) {
                    const int lrow = g * 16 + r + half * 8;
                    s_m[wave * M_ROWS + lrow] = m;
                    s_s[wave * M_ROWS + lrow] = s;
                }
            }
        }
        __syncthreads();
        if (tid < M_ROWS) {
            float m = s_m[tid], s = s_s[tid];
            #pragma unroll
            for (int w = 1; w < WAVES; ++w) {
                float mo = s_m[w * M_ROWS + tid], so = s_s[w * M_ROWS + tid];
                float mn = fmaxf(m, mo);
                s = s * __expf(m - mn) + so * __expf(mo - mn);
                m = mn;
            }
            const size_t pidx = (size_t)(seg * CHUNKS + chunk) * N_TOK + rowBase + tid;
            pm[pidx] = m;
            ps[pidx] = s;
        }
        __syncthreads();
    }
}

__global__ void adaptive_lsm_finalize(const int* __restrict__ target,
                                      const float* __restrict__ pm,
                                      const float* __restrict__ ps,
                                      const float* __restrict__ gtgt,
                                      const float* __restrict__ gclus,
                                      float* __restrict__ out, int n) {
    int row = blockIdx.x * blockDim.x + threadIdx.x;
    if (row >= n) return;
    float lse[3];
    for (int seg = 0; seg < 3; ++seg) {
        float m = NEG_INF, s = 0.f;
        for (int c = 0; c < CHUNKS; ++c) {
            size_t idx = (size_t)(seg * CHUNKS + c) * N_TOK + row;
            float mo = pm[idx], so = ps[idx];
            float mn = fmaxf(m, mo);
            s = s * __expf(m - mn) + so * __expf(mo - mn);
            m = mn;
        }
        lse[seg] = m + __logf(s);
    }
    const int t = target[row];
    float nll;
    if (t < 20000)      nll = lse[0] - gtgt[row];
    else if (t < 40000) nll = (lse[0] - gclus[N_TOK + row]) + (lse[1] - gtgt[N_TOK + row]);
    else                nll = (lse[0] - gclus[row])         + (lse[2] - gtgt[2 * N_TOK + row]);
    out[row] = nll;
}

extern "C" void kernel_launch(void* const* d_in, const int* in_sizes, int n_in,
                              void* d_out, int out_size, void* d_ws, size_t ws_size,
                              hipStream_t stream) {
    const float* hidden = (const float*)d_in[0];
    const int*   target = (const int*)d_in[1];
    const float* weight = (const float*)d_in[2];
    const float* bias   = (const float*)d_in[3];
    const float* cw     = (const float*)d_in[4];
    const float* cb     = (const float*)d_in[5];
    float* out = (float*)d_out;

    char* ws   = (char*)d_ws;
    u16*  whi  = (u16*)(ws + OFF_WHI);
    u16*  wlo  = (u16*)(ws + OFF_WLO);
    float* pm  = (float*)(ws + OFF_PM);
    float* ps  = (float*)(ws + OFF_PS);
    float* gtgt  = (float*)(ws + OFF_TGT);
    float* gclus = (float*)(ws + OFF_CLUS);

    convert_weight_split<<<2048, 256, 0, stream>>>(weight, cw, whi, wlo);
    init_gather<<<(3 * N_TOK + 255) / 256, 256, 0, stream>>>(gtgt, gclus);

    const size_t shmem = (size_t)M_ROWS * A_STRIDE * 2 * sizeof(u16)
                       + M_ROWS * sizeof(int)
                       + (size_t)2 * WAVES * M_ROWS * sizeof(float);   // ~262 KB
    hipFuncSetAttribute((const void*)adaptive_lsm_gemm,
                        hipFuncAttributeMaxDynamicSharedMemorySize, (int)shmem);
    dim3 grid(N_TOK / M_ROWS, CHUNKS);
    adaptive_lsm_gemm<<<grid, 256, shmem, stream>>>(hidden, target, bias, cb,
                                                    whi, wlo, pm, ps, gtgt, gclus);
    adaptive_lsm_finalize<<<(N_TOK + 255) / 256, 256, 0, stream>>>(
        target, pm, ps, gtgt, gclus, out, N_TOK);
}
